// HeteroGINE_11527692223014
// MI455X (gfx1250) — matrix operand order repaired and checked by reading, not compiled, not run on hardware
//
#include <hip/hip_runtime.h>

#define NN 100000
#define EE 500000
#define HH 128

typedef __bf16 bf16_t;
typedef __attribute__((ext_vector_type(16))) __bf16 v16bf;
typedef __attribute__((ext_vector_type(8)))  float  v8f;

union FragB { uint4 u[2]; v16bf v; };   // 32 bytes = 16 bf16 = one WMMA A/B fragment per lane

static constexpr int LDH = 136;   // padded LDS stride (halves): 272B, multiple of 16B
static constexpr int LDO = 132;   // padded LDS stride (floats) for output tile

// ---------------------------------------------------------------------------
// zero fill (float4)
// ---------------------------------------------------------------------------
__global__ __launch_bounds__(256) void zero_kernel(float4* __restrict__ p, int n4) {
  int i = blockIdx.x * 256 + threadIdx.x;
  if (i < n4) p[i] = make_float4(0.f, 0.f, 0.f, 0.f);
}

// ---------------------------------------------------------------------------
// GINE edge message + scatter-add:
//   agg[dst] += relu(x_src[src] + ea*We + be)
// One wave32 per edge; lane l owns columns [4l, 4l+4).
// ---------------------------------------------------------------------------
__global__ __launch_bounds__(256) void edge_kernel(
    const float* __restrict__ xsrc, const int* __restrict__ ei,
    const float* __restrict__ ea, const float* __restrict__ we,
    const float* __restrict__ be, float* __restrict__ agg) {
  const int lane = threadIdx.x & 31;
  const int gw   = (blockIdx.x * 256 + threadIdx.x) >> 5;
  const int nw   = (gridDim.x * 256) >> 5;
  const float4 w4 = ((const float4*)we)[lane];
  const float4 b4 = ((const float4*)be)[lane];
  for (int e = gw; e < EE; e += nw) {
    const int   s = ei[e];
    const int   d = ei[EE + e];
    const float a = ea[e];
    const float4 x = *(const float4*)(xsrc + (size_t)s * HH + lane * 4);
    const float m0 = fmaxf(fmaf(a, w4.x, x.x) + b4.x, 0.f);
    const float m1 = fmaxf(fmaf(a, w4.y, x.y) + b4.y, 0.f);
    const float m2 = fmaxf(fmaf(a, w4.z, x.z) + b4.z, 0.f);
    const float m3 = fmaxf(fmaf(a, w4.w, x.w) + b4.w, 0.f);
    float* p = agg + (size_t)d * HH + lane * 4;
    atomicAdd(p + 0, m0);
    atomicAdd(p + 1, m1);
    atomicAdd(p + 2, m2);
    atomicAdd(p + 3, m3);
  }
}

// ---------------------------------------------------------------------------
// Fused GEMM + epilogue:
//   Z = (x0 [+ x1]) @ W + bias          (bf16 WMMA, f32 accumulate)
//   Y = do_ln ? relu(LN(Z)*gamma+beta) : relu(Z)
// Block = 256 threads (8 waves), 64 rows x 128 cols of output.
// Safe in-place (Y may alias x0/x1): each block reads its 64 rows into LDS
// before writing the same 64 rows.
// ---------------------------------------------------------------------------
__global__ __launch_bounds__(256) void gemm128_kernel(
    const float* __restrict__ x0, const float* __restrict__ x1,
    const float* __restrict__ W,  const float* __restrict__ bias,
    const float* __restrict__ gamma, const float* __restrict__ beta,
    float* __restrict__ Y, int nrows, int do_ln) {
  __shared__ union SM {
    struct { bf16_t X[64 * LDH]; bf16_t Wt[HH * LDH]; } in;   // ~52 KB
    struct { float O[64 * LDO]; float mean[64]; float rstd[64]; } out;
  } sm;

  const int tid     = threadIdx.x;
  const int rowBase = blockIdx.x * 64;

  // Stage X tile (64x128) -> bf16 LDS, fusing optional x1 add.
#pragma unroll
  for (int i = 0; i < 32; ++i) {
    const int idx = tid + i * 256;            // 0..8191
    const int r = idx >> 7, k = idx & 127;
    const int gr = rowBase + r;
    float v = 0.f;
    if (gr < nrows) {
      v = x0[(size_t)gr * HH + k];
      if (x1) v += x1[(size_t)gr * HH + k];
    }
    sm.in.X[r * LDH + k] = (bf16_t)v;
  }
  // Stage W (128x128, [k][n] row-major) transposed -> Wt[n][k] bf16 LDS.
#pragma unroll
  for (int i = 0; i < 64; ++i) {
    const int idx = tid + i * 256;            // 0..16383
    const int k = idx >> 7, n = idx & 127;
    sm.in.Wt[n * LDH + k] = (bf16_t)W[idx];
  }
  __syncthreads();

  const int wave = tid >> 5;
  const int lane = tid & 31;
  const int rt   = wave >> 1;        // row tile 0..3  (rows rt*16..)
  const int cg   = wave & 1;         // col group 0..1 (cols cg*64..)
  const int lrow = lane & 15;
  const int lg   = lane >> 4;

  v8f acc[4] = {};
  const bf16_t* Xrow = &sm.in.X[(rt * 16 + lrow) * LDH];

#pragma unroll
  for (int kc = 0; kc < 4; ++kc) {           // K = 128 in chunks of 32
    const int kb = kc * 32;
    // A fragment (16x32 bf16): lanes 0-15 hold K {0..7,16..23}, 16-31 hold {8..15,24..31}
    FragB fa;
    fa.u[0] = *(const uint4*)(Xrow + kb + lg * 8);
    fa.u[1] = *(const uint4*)(Xrow + kb + lg * 8 + 16);
#pragma unroll
    for (int t = 0; t < 4; ++t) {
      // B fragment (32x16 bf16): lane L -> column n = L&15, lanes 0-15 hold K {0..7,16..23}, 16-31 hold {8..15,24..31}
      const bf16_t* wp = &sm.in.Wt[(cg * 64 + t * 16 + lrow) * LDH + kb + lg * 8];
      FragB fb;
      fb.u[0] = *(const uint4*)(wp);
      fb.u[1] = *(const uint4*)(wp + 16);
      acc[t] = __builtin_amdgcn_wmma_f32_16x16x32_bf16(
          false, fa.v, false, fb.v, (short)0, acc[t], false, false);
    }
  }
  __syncthreads();   // done reading in-tiles; smem becomes the out-tile

  // Spill accumulators per documented C layout: VGPR r -> M=r (lanes 0-15), M=r+8 (16-31)
  const int mbase = rt * 16 + lg * 8;
#pragma unroll
  for (int t = 0; t < 4; ++t)
#pragma unroll
    for (int r = 0; r < 8; ++r)
      sm.out.O[(mbase + r) * LDO + cg * 64 + t * 16 + lrow] = acc[t][r];
  __syncthreads();

  // LayerNorm statistics (bias folded in), one thread per row.
  if (do_ln && tid < 64) {
    float s = 0.f, s2 = 0.f;
#pragma unroll 4
    for (int k = 0; k < HH; ++k) {
      const float z = sm.out.O[tid * LDO + k] + bias[k];
      s += z; s2 += z * z;
    }
    const float m = s * (1.f / HH);
    const float var = s2 * (1.f / HH) - m * m;
    sm.out.mean[tid] = m;
    sm.out.rstd[tid] = rsqrtf(var + 1e-5f);
  }
  __syncthreads();

  // Apply epilogue + coalesced store.
#pragma unroll
  for (int i = 0; i < 32; ++i) {
    const int idx = tid + i * 256;
    const int r = idx >> 7, k = idx & 127;
    const int gr = rowBase + r;
    if (gr < nrows) {
      const float z = sm.out.O[r * LDO + k] + bias[k];
      float o;
      if (do_ln)
        o = (z - sm.out.mean[r]) * sm.out.rstd[r] * gamma[k] + beta[k];
      else
        o = z;
      Y[(size_t)gr * HH + k] = fmaxf(o, 0.f);
    }
  }
}

// ---------------------------------------------------------------------------
// Host-side orchestration
// ---------------------------------------------------------------------------
extern "C" void kernel_launch(void* const* d_in, const int* in_sizes, int n_in,
                              void* d_out, int out_size, void* d_ws, size_t ws_size,
                              hipStream_t stream) {
  (void)in_sizes; (void)n_in; (void)out_size; (void)ws_size;

  const size_t NH = (size_t)NN * HH;
  const float* x_a = (const float*)d_in[0];
  const float* x_b = (const float*)d_in[1];

  float* A0  = (float*)d_ws;          // scratch aggregation / tmp buffer (51.2 MB)
  float* O_a = (float*)d_out;         // out_a (also used as intermediate)
  float* O_b = O_a + NH;              // out_b (also used as intermediate)

  const int ZB = (int)((NH / 4 + 255) / 256);   // zero-fill blocks
  const int EB = (EE + 7) / 8;                  // 8 waves/block, 1 edge/wave
  const int GB = (NN + 63) / 64;                // gemm blocks

  auto F = [&](int i) { return (const float*)d_in[i]; };

  auto gemm = [&](const float* p0, const float* p1, int wi, int bi, int gi, int ci,
                  float* y, int ln) {
    gemm128_kernel<<<GB, 256, 0, stream>>>(
        p0, p1, F(wi), F(bi),
        gi >= 0 ? F(gi) : nullptr, ci >= 0 ? F(ci) : nullptr, y, NN, ln);
  };

  // One GINE conv: agg=A0 (zero + scatter), then 2x fused GEMM+LN+ReLU.
  auto conv = [&](const float* xsrc, const float* xdst, int s, float* tmp, float* out) {
    zero_kernel<<<ZB, 256, 0, stream>>>((float4*)A0, (int)(NH / 4));
    edge_kernel<<<EB, 256, 0, stream>>>(xsrc, (const int*)d_in[s], F(s + 1), F(s + 2),
                                        F(s + 3), A0);
    gemm(A0,  xdst,    s + 4, s + 5, s + 6,  s + 7,  tmp, 1);   // LN1
    gemm(tmp, nullptr, s + 8, s + 9, s + 10, s + 11, out, 1);   // LN2
  };

  // Input index blocks: ab=2, ba=14, aa=26, post_a=38, post_b=44
  conv(x_a, x_a, 26, O_a, O_b);                 // gine_aa -> O_b
  conv(x_b, x_a, 14, O_a, O_a);                 // gine_ba -> O_a (in-place 2nd gemm)
  gemm(O_a, O_b,     38, 39, 40, 41, O_a, 1);   // post_a hidden (sum fused, in-place)
  gemm(O_a, nullptr, 42, 43, -1, -1, O_a, 0);   // out_a = relu(.@Wp2+bp2)
  conv(x_a, x_b, 2, O_b, O_b);                  // gine_ab -> O_b
  gemm(O_b, nullptr, 44, 45, 46, 47, O_b, 1);   // post_b hidden
  gemm(O_b, nullptr, 48, 49, -1, -1, O_b, 0);   // out_b
}
